// E_GCL_LN_9414568313009
// MI455X (gfx1250) — matrix operand order repaired and checked by reading, not compiled, not run on hardware
//
#include <hip/hip_runtime.h>
#include <hip/hip_bf16.h>
#include <math.h>

typedef __attribute__((ext_vector_type(16))) __bf16 v16bf;
typedef __attribute__((ext_vector_type(8)))  __bf16 v8bf;
typedef __attribute__((ext_vector_type(4)))  __bf16 v4bf;
typedef __attribute__((ext_vector_type(8)))  float  v8f;

#define HDIM 128
#define TILE 64          // edges/nodes per block
#define NTHREADS 256     // 8 wave32
#define EPS_LN 1e-5f

__device__ __forceinline__ float silu_f(float x) { return x / (1.0f + __expf(-x)); }

// ---------------------------------------------------------------------------
// Stage a [K x 128] f32 weight matrix from global into LDS as bf16 in exact
// B-fragment order: dst[((kt*8 + nt)*32 + lane)*16 + e] where
//   kt = k/32, e = k%16, lane = 16*((k%32)/16) + (n%16), nt = n/16.
// Each lane's 16 fragment elements land contiguous (32 B) -> 2x ds_load_b128.
// ---------------------------------------------------------------------------
__device__ __forceinline__ void stage_weights_swz(int tid,
                                                  const float* __restrict__ W,
                                                  int K, __bf16* __restrict__ dst) {
  for (int i = tid * 4; i < K * HDIM; i += NTHREADS * 4) {
    float4 w4 = *(const float4*)(W + i);     // coalesced, L2-resident
    const float wv[4] = {w4.x, w4.y, w4.z, w4.w};
    int k  = i >> 7;
    int n0 = i & 127;
    int kt = k >> 5;
    int kr = k & 31;
    int lh = (kr >> 4) << 4;                 // lane-half base: 0 or 16
    int e  = kr & 15;
#pragma unroll
    for (int u = 0; u < 4; ++u) {
      int n    = n0 + u;
      int lane = lh + (n & 15);
      int nt   = n >> 4;
      dst[((((kt << 3) + nt) << 5) + lane) * 16 + e] = (__bf16)wv[u];
    }
  }
}

// ---------------------------------------------------------------------------
// 64xK @ Kx128 GEMM, bf16 in LDS, f32 WMMA accumulate.
// Wave w owns M-tile (w&3), N-tiles [(w>>2)*4 .. +4). kTiles = K/32.
// A fragment: lane l -> row M=l&15; two aligned 16B loads (K 16*kh + 0..7 / 8..15).
// B fragment: contiguous 32B per lane from the swizzled weight buffer.
// ---------------------------------------------------------------------------
__device__ __forceinline__ void gemm_tile(int tid, const __bf16* __restrict__ A,
                                          int astride, int kTiles,
                                          const __bf16* __restrict__ Wswz,
                                          v8f acc[4]) {
  const int lane = tid & 31;
  const int wave = tid >> 5;
  const int mt   = wave & 3;
  const int nts  = (wave >> 2) << 2;
  const int mrow = (mt << 4) + (lane & 15);
  const int kh   = lane >> 4;
  v8f z = {};
  acc[0] = z; acc[1] = z; acc[2] = z; acc[3] = z;
  const __bf16* arow = A + mrow * astride + (kh << 4);
  for (int kt = 0; kt < kTiles; ++kt) {
    const __bf16* ap = arow + (kt << 5);
    v8bf alo = *(const v8bf*)ap;             // K = kbase + 16*kh + 0..7
    v8bf ahi = *(const v8bf*)(ap + 8);       // K = kbase + 16*kh + 8..15
    v16bf afrag = __builtin_shufflevector(alo, ahi,
        0, 1, 2, 3, 4, 5, 6, 7, 8, 9, 10, 11, 12, 13, 14, 15);
#pragma unroll
    for (int j = 0; j < 4; ++j) {
      const __bf16* bp = Wswz + ((((kt << 3) + nts + j) << 5) + lane) * 16;
      v8bf blo = *(const v8bf*)bp;
      v8bf bhi = *(const v8bf*)(bp + 8);
      v16bf bfrag = __builtin_shufflevector(blo, bhi,
          0, 1, 2, 3, 4, 5, 6, 7, 8, 9, 10, 11, 12, 13, 14, 15);
      acc[j] = __builtin_amdgcn_wmma_f32_16x16x32_bf16(
          false, afrag, false, bfrag, (short)0, acc[j], false, false);
    }
  }
}

// Accumulator -> LDS f32 scratch (adds bias). C/D layout: lane l -> N=l&15,
// element r -> M = 8*(l>>4)+r inside the 16-row tile.
__device__ __forceinline__ void writeback_tile(int tid, const v8f acc[4],
                                               float* __restrict__ scr,
                                               const float* __restrict__ biasv) {
  const int lane = tid & 31;
  const int wave = tid >> 5;
  const int mt   = wave & 3;
  const int nts  = (wave >> 2) << 2;
#pragma unroll
  for (int j = 0; j < 4; ++j)
#pragma unroll
    for (int r = 0; r < 8; ++r) {
      int m = (mt << 4) + ((lane >> 4) << 3) + r;
      int n = ((nts + j) << 4) + (lane & 15);
      scr[m * HDIM + n] = acc[j][r] + biasv[n];
    }
}

// Row-wise LayerNorm + SiLU over [64][128] f32 LDS tile; 4 threads per row,
// float4 vectorized. Optionally emits bf16 copy and/or rewrites f32 in place.
__device__ __forceinline__ void layernorm_silu(int tid, float* __restrict__ scr,
                                               const float* __restrict__ gv,
                                               const float* __restrict__ bv,
                                               float* __restrict__ lnS,
                                               float* __restrict__ lnS2,
                                               __bf16* __restrict__ outBf,
                                               bool writeF32) {
  const int r = tid >> 2, q = tid & 3;
  float s = 0.f, s2 = 0.f;
  for (int c = q * 32; c < q * 32 + 32; c += 4) {
    float4 x = *(const float4*)(scr + r * HDIM + c);
    s  += x.x + x.y + x.z + x.w;
    s2 += x.x * x.x + x.y * x.y + x.z * x.z + x.w * x.w;
  }
  lnS[r * 4 + q] = s; lnS2[r * 4 + q] = s2;
  __syncthreads();
  float mu  = (lnS[r*4+0] + lnS[r*4+1] + lnS[r*4+2] + lnS[r*4+3]) * (1.f/128.f);
  float var = (lnS2[r*4+0] + lnS2[r*4+1] + lnS2[r*4+2] + lnS2[r*4+3]) * (1.f/128.f) - mu*mu;
  float rs  = rsqrtf(var + EPS_LN);
  for (int c = q * 32; c < q * 32 + 32; c += 4) {
    float4 x = *(const float4*)(scr + r * HDIM + c);
    float y0 = silu_f((x.x - mu) * rs * gv[c+0] + bv[c+0]);
    float y1 = silu_f((x.y - mu) * rs * gv[c+1] + bv[c+1]);
    float y2 = silu_f((x.z - mu) * rs * gv[c+2] + bv[c+2]);
    float y3 = silu_f((x.w - mu) * rs * gv[c+3] + bv[c+3]);
    if (outBf) {
      v4bf t; t[0] = (__bf16)y0; t[1] = (__bf16)y1;
              t[2] = (__bf16)y2; t[3] = (__bf16)y3;
      *(v4bf*)(outBf + r * HDIM + c) = t;
    }
    if (writeF32) {
      float4 y; y.x = y0; y.y = y1; y.z = y2; y.w = y3;
      *(float4*)(scr + r * HDIM + c) = y;
    }
  }
  __syncthreads();
}

// ---------------------------------------------------------------------------
// d_out[0 .. N*128) = 0 (agg accumulator, later overwritten by node output)
// d_out[N*128 .. N*131) = coord (coord_out base for scatter)
// ---------------------------------------------------------------------------
__global__ void init_out_kernel(const float* __restrict__ coord,
                                float* __restrict__ out, int N) {
  int i = blockIdx.x * blockDim.x + threadIdx.x;
  int nh = N * HDIM;
  if (i < nh) out[i] = 0.0f;
  else if (i < nh + N * 3) out[i] = coord[i - nh];
}

// ---------------------------------------------------------------------------
// Fused edge kernel: gather -> (258->128 GEMM,LN,SiLU) -> (128->128,LN,SiLU)
//   -> agg scatter -> (128->128,LN,SiLU) -> dot cW2 -> coord scatter.
// ---------------------------------------------------------------------------
__global__ __launch_bounds__(NTHREADS) void edge_kernel(
    const float* __restrict__ h, const long long* __restrict__ eidx,
    const float* __restrict__ coord, const float* __restrict__ eattr,
    const float* __restrict__ eW1, const float* __restrict__ eb1,
    const float* __restrict__ eg1, const float* __restrict__ ebe1,
    const float* __restrict__ eW2, const float* __restrict__ eb2,
    const float* __restrict__ eg2, const float* __restrict__ ebe2,
    const float* __restrict__ cW1, const float* __restrict__ cb1,
    const float* __restrict__ cg1, const float* __restrict__ cbe1,
    const float* __restrict__ cW2,
    float* __restrict__ agg, float* __restrict__ coordOut, int E) {
  __shared__ __bf16 wB[256 * HDIM];                       // 64 KB, swizzled
  __shared__ union { __bf16 inp[TILE][256]; float scr[TILE][HDIM]; } U;  // 32 KB
  __shared__ __bf16 actA[TILE][HDIM];                     // 16 KB
  __shared__ __bf16 actB[TILE][HDIM];                     // 16 KB
  __shared__ float wTail[2][HDIM];                        // e_W1 rows 256,257
  __shared__ float biasv[HDIM], gammav[HDIM], betav[HDIM], cw2[HDIM];
  __shared__ float radS[TILE], eaS[TILE], cdf[TILE][3];
  __shared__ long long rowS[TILE], colS[TILE];
  __shared__ float lnS[TILE * 4], lnS2[TILE * 4];

  const int tid = threadIdx.x;
  const int e0  = blockIdx.x * TILE;

  // ---- stage geometry / indices / phase-1 weights -------------------------
  for (int r = tid; r < TILE; r += NTHREADS) {
    int e = e0 + r; int ec = (e < E) ? e : (E - 1);
    long long ri = eidx[ec], ci = eidx[(long long)E + ec];
    rowS[r] = ri; colS[r] = ci;
    float dx = coord[ri*3+0] - coord[ci*3+0];
    float dy = coord[ri*3+1] - coord[ci*3+1];
    float dz = coord[ri*3+2] - coord[ci*3+2];
    float rad = dx*dx + dy*dy + dz*dz;
    radS[r] = rad;
    float inv = 1.0f / (sqrtf(rad + 1e-8f) + 1.0f);
    cdf[r][0] = dx * inv; cdf[r][1] = dy * inv; cdf[r][2] = dz * inv;
    eaS[r] = eattr[ec];
  }
  for (int i = tid; i < HDIM; i += NTHREADS) {
    wTail[0][i] = eW1[256 * HDIM + i];
    wTail[1][i] = eW1[257 * HDIM + i];
    biasv[i] = eb1[i]; gammav[i] = eg1[i]; betav[i] = ebe1[i];
    cw2[i] = cW2[i];
  }
  stage_weights_swz(tid, eW1, 256, wB);
  __syncthreads();

  // gather h[row] | h[col] into bf16 input tile (float4 coalesced rows)
  for (int i = tid * 4; i < TILE * 256; i += NTHREADS * 4) {
    int r = i >> 8, c = i & 255;
    const float* src = (c < HDIM) ? (h + rowS[r] * HDIM + c)
                                  : (h + colS[r] * HDIM + (c - HDIM));
    float4 x = *(const float4*)src;
    v4bf t; t[0] = (__bf16)x.x; t[1] = (__bf16)x.y;
            t[2] = (__bf16)x.z; t[3] = (__bf16)x.w;
    *(v4bf*)(&U.inp[r][c]) = t;
  }
  __syncthreads();

  // ---- GEMM1: [64x256] @ [256x128] + rank-2 tail (radial, edge_attr) -----
  v8f acc[4];
  gemm_tile(tid, &U.inp[0][0], 256, 8, wB, acc);
  __syncthreads();                       // all reads of inp/wB done
  {
    const int lane = tid & 31, wave = tid >> 5;
    const int mt = wave & 3, nts = (wave >> 2) << 2;
#pragma unroll
    for (int j = 0; j < 4; ++j)
#pragma unroll
      for (int r = 0; r < 8; ++r) {
        int m = (mt << 4) + ((lane >> 4) << 3) + r;
        int n = ((nts + j) << 4) + (lane & 15);
        U.scr[m][n] = acc[j][r] + biasv[n]
                    + radS[m] * wTail[0][n] + eaS[m] * wTail[1][n];
      }
  }
  stage_weights_swz(tid, eW2, 128, wB);
  __syncthreads();
  layernorm_silu(tid, &U.scr[0][0], gammav, betav, lnS, lnS2, &actA[0][0], false);

  for (int i = tid; i < HDIM; i += NTHREADS) {
    biasv[i] = eb2[i]; gammav[i] = eg2[i]; betav[i] = ebe2[i];
  }
  __syncthreads();

  // ---- GEMM2: edge_feat = silu(LN(actA @ eW2 + b2)) ----------------------
  gemm_tile(tid, &actA[0][0], HDIM, 4, wB, acc);
  __syncthreads();
  writeback_tile(tid, acc, &U.scr[0][0], biasv);
  stage_weights_swz(tid, cW1, 128, wB);
  __syncthreads();
  layernorm_silu(tid, &U.scr[0][0], gammav, betav, lnS, lnS2, &actB[0][0], true);

  // scatter agg += edge_feat (f32, from scratch) + load phase-3 LN params
  for (int i = tid; i < TILE * HDIM; i += NTHREADS) {
    int r = i >> 7, c = i & 127;
    if (e0 + r < E) atomicAdd(&agg[rowS[r] * HDIM + c], U.scr[r][c]);
  }
  for (int i = tid; i < HDIM; i += NTHREADS) {
    biasv[i] = cb1[i]; gammav[i] = cg1[i]; betav[i] = cbe1[i];
  }
  __syncthreads();

  // ---- GEMM3: coord MLP hidden, then dot with cW2 ------------------------
  gemm_tile(tid, &actB[0][0], HDIM, 4, wB, acc);
  __syncthreads();
  writeback_tile(tid, acc, &U.scr[0][0], biasv);
  __syncthreads();
  layernorm_silu(tid, &U.scr[0][0], gammav, betav, lnS, lnS2, nullptr, true);
  {
    const int r = tid >> 2, q = tid & 3;
    float s = 0.f;
    for (int c = q * 32; c < q * 32 + 32; c += 4) {
      float4 x = *(const float4*)(&U.scr[r][c]);
      float4 w = *(const float4*)(&cw2[c]);
      s += x.x * w.x + x.y * w.y + x.z * w.z + x.w * w.w;
    }
    lnS[r * 4 + q] = s;
  }
  __syncthreads();
  if (tid < TILE && (e0 + tid) < E) {
    float cm = lnS[tid*4+0] + lnS[tid*4+1] + lnS[tid*4+2] + lnS[tid*4+3];
    long long ri = rowS[tid];
    atomicAdd(&coordOut[ri * 3 + 0], cdf[tid][0] * cm);
    atomicAdd(&coordOut[ri * 3 + 1], cdf[tid][1] * cm);
    atomicAdd(&coordOut[ri * 3 + 2], cdf[tid][2] * cm);
  }
}

// ---------------------------------------------------------------------------
// Fused node kernel: out = h + nW2 @ silu(LN([h|agg] @ nW1 + b1)) + b2.
// agg lives in the out buffer; each block reads its rows then overwrites them.
// ---------------------------------------------------------------------------
__global__ __launch_bounds__(NTHREADS) void node_kernel(
    const float* __restrict__ h,
    const float* __restrict__ nW1, const float* __restrict__ nb1,
    const float* __restrict__ ng1, const float* __restrict__ nbe1,
    const float* __restrict__ nW2, const float* __restrict__ nb2,
    float* __restrict__ outBuf, int N) {
  __shared__ __bf16 wB[256 * HDIM];
  __shared__ union { __bf16 inp[TILE][256]; float scr[TILE][HDIM]; } U;
  __shared__ __bf16 actA[TILE][HDIM];
  __shared__ float biasv[HDIM], gammav[HDIM], betav[HDIM];
  __shared__ float lnS[TILE * 4], lnS2[TILE * 4];

  const int tid = threadIdx.x;
  const int r0  = blockIdx.x * TILE;

  for (int i = tid * 4; i < TILE * 256; i += NTHREADS * 4) {
    int r = i >> 8, c = i & 255;
    int rc = r0 + r; if (rc >= N) rc = N - 1;
    const float* src = (c < HDIM) ? (h + (long long)rc * HDIM + c)
                                  : (outBuf + (long long)rc * HDIM + (c - HDIM));
    float4 x = *(const float4*)src;
    v4bf t; t[0] = (__bf16)x.x; t[1] = (__bf16)x.y;
            t[2] = (__bf16)x.z; t[3] = (__bf16)x.w;
    *(v4bf*)(&U.inp[r][c]) = t;
  }
  for (int i = tid; i < HDIM; i += NTHREADS) {
    biasv[i] = nb1[i]; gammav[i] = ng1[i]; betav[i] = nbe1[i];
  }
  stage_weights_swz(tid, nW1, 256, wB);
  __syncthreads();

  v8f acc[4];
  gemm_tile(tid, &U.inp[0][0], 256, 8, wB, acc);
  __syncthreads();
  writeback_tile(tid, acc, &U.scr[0][0], biasv);
  stage_weights_swz(tid, nW2, 128, wB);
  __syncthreads();
  layernorm_silu(tid, &U.scr[0][0], gammav, betav, lnS, lnS2, &actA[0][0], false);
  for (int i = tid; i < HDIM; i += NTHREADS) biasv[i] = nb2[i];
  __syncthreads();

  gemm_tile(tid, &actA[0][0], HDIM, 4, wB, acc);
  __syncthreads();
  writeback_tile(tid, acc, &U.scr[0][0], biasv);
  __syncthreads();

  for (int i = tid * 4; i < TILE * HDIM; i += NTHREADS * 4) {
    int r = i >> 7, c = i & 127;
    int rc = r0 + r;
    if (rc < N) {
      float4 sv = *(const float4*)(&U.scr[r][c]);
      float4 hv = *(const float4*)(h + (long long)rc * HDIM + c);
      float4 o; o.x = sv.x + hv.x; o.y = sv.y + hv.y;
                o.z = sv.z + hv.z; o.w = sv.w + hv.w;
      *(float4*)(outBuf + (long long)rc * HDIM + c) = o;
    }
  }
}

// ---------------------------------------------------------------------------
extern "C" void kernel_launch(void* const* d_in, const int* in_sizes, int n_in,
                              void* d_out, int out_size, void* d_ws, size_t ws_size,
                              hipStream_t stream) {
  const float*     h     = (const float*)d_in[0];
  const long long* eidx  = (const long long*)d_in[1];
  const float*     coord = (const float*)d_in[2];
  const float*     eattr = (const float*)d_in[3];
  const float *eW1 = (const float*)d_in[4],  *eb1 = (const float*)d_in[5];
  const float *eg1 = (const float*)d_in[6],  *ebe1 = (const float*)d_in[7];
  const float *eW2 = (const float*)d_in[8],  *eb2 = (const float*)d_in[9];
  const float *eg2 = (const float*)d_in[10], *ebe2 = (const float*)d_in[11];
  const float *nW1 = (const float*)d_in[12], *nb1 = (const float*)d_in[13];
  const float *ng1 = (const float*)d_in[14], *nbe1 = (const float*)d_in[15];
  const float *nW2 = (const float*)d_in[16], *nb2 = (const float*)d_in[17];
  const float *cW1 = (const float*)d_in[18], *cb1 = (const float*)d_in[19];
  const float *cg1 = (const float*)d_in[20], *cbe1 = (const float*)d_in[21];
  const float *cW2 = (const float*)d_in[22];

  const int N = in_sizes[0] / HDIM;   // 25000
  const int E = in_sizes[1] / 2;      // 400000

  float* out      = (float*)d_out;             // [N,128]: agg then final out
  float* coordOut = out + (size_t)N * HDIM;    // [N,3]

  int initTotal = N * (HDIM + 3);
  init_out_kernel<<<(initTotal + 255) / 256, 256, 0, stream>>>(coord, out, N);

  edge_kernel<<<(E + TILE - 1) / TILE, NTHREADS, 0, stream>>>(
      h, eidx, coord, eattr,
      eW1, eb1, eg1, ebe1, eW2, eb2, eg2, ebe2,
      cW1, cb1, cg1, cbe1, cW2, out, coordOut, E);

  node_kernel<<<(N + TILE - 1) / TILE, NTHREADS, 0, stream>>>(
      h, nW1, nb1, ng1, nbe1, nW2, nb2, out, N);
}
